// TripleLatentLayer_69123203662024
// MI455X (gfx1250) — matrix operand order repaired and checked by reading, not compiled, not run on hardware
//
#include <hip/hip_runtime.h>
#include <math.h>

#define B_     8
#define T_     2048
#define DM_    1024
#define H_     8
#define SD_    64
#define INNER_ 512
#define HID_   4096
#define M_     (B_ * T_)   // 16384

typedef __attribute__((ext_vector_type(16))) __bf16 v16bf;
typedef __attribute__((ext_vector_type(8)))  __bf16 v8bf;
typedef __attribute__((ext_vector_type(8)))  float  v8f;
typedef __attribute__((ext_vector_type(4)))  int    v4i;

static __device__ __forceinline__ float  bf2f(__bf16 x) { return (float)x; }
static __device__ __forceinline__ __bf16 f2bf(float x)  { return (__bf16)x; }

// gfx1250 async global->LDS copy (ASYNCcnt-tracked). Device pass only.
#if defined(__gfx1250__) && __has_builtin(__builtin_amdgcn_global_load_async_to_lds_b128)
#define USE_ASYNC_LDS 1
#else
#define USE_ASYNC_LDS 0
#endif

#if USE_ASYNC_LDS
// Builtin signature (from clang diagnostics): (v4i AS1*, v4i AS3*, imm, imm)
#define ASYNC_COPY16(gsrc, ldst)                                               \
    __builtin_amdgcn_global_load_async_to_lds_b128(                            \
        (__attribute__((address_space(1))) v4i*)(v4i*)(void*)(gsrc),           \
        (__attribute__((address_space(3))) v4i*)(v4i*)(void*)(ldst), 0, 0)
static __device__ __forceinline__ void wait_async_lds() {
#if __has_builtin(__builtin_amdgcn_s_wait_asynccnt)
    __builtin_amdgcn_s_wait_asynccnt(0);
#else
    asm volatile("s_wait_asynccnt 0x0" ::: "memory");
#endif
}
#endif

// ---------------------------------------------------------------------------
// f32 -> bf16 conversion (grid-stride)
// ---------------------------------------------------------------------------
__global__ __launch_bounds__(256)
void f32_to_bf16_kernel(const float* __restrict__ in, __bf16* __restrict__ out, long long n) {
    long long i = (long long)blockIdx.x * blockDim.x + threadIdx.x;
    long long stride = (long long)gridDim.x * blockDim.x;
    for (; i < n; i += stride) out[i] = f2bf(in[i]);
}

// ---------------------------------------------------------------------------
// f32 [K][N] -> bf16 transposed [N][K]  (32x32 LDS tiles; K,N % 32 == 0)
// Pre-transposing weights lets the GEMM stage B with vector ds_store_b128.
// ---------------------------------------------------------------------------
__global__ __launch_bounds__(256)
void transpose_to_bf16_kernel(const float* __restrict__ in, __bf16* __restrict__ out,
                              int K, int N) {
    __shared__ float tile[32][33];
    const int kb = blockIdx.x * 32;
    const int nb = blockIdx.y * 32;
    const int tx = threadIdx.x & 31;
    const int ty = threadIdx.x >> 5;   // 0..7
#pragma unroll
    for (int i = 0; i < 32; i += 8)
        tile[ty + i][tx] = in[(size_t)(kb + ty + i) * N + (nb + tx)];
    __syncthreads();
#pragma unroll
    for (int i = 0; i < 32; i += 8)
        out[(size_t)(nb + ty + i) * K + (kb + tx)] = f2bf(tile[tx][ty + i]);
}

// ---------------------------------------------------------------------------
// Row layernorm (row length DM_=1024), bf16 output
// ---------------------------------------------------------------------------
__global__ __launch_bounds__(256)
void layernorm_bf16_kernel(const float* __restrict__ x, const float* __restrict__ g,
                           const float* __restrict__ b, __bf16* __restrict__ out) {
    __shared__ float ssum[256];
    __shared__ float ssq[256];
    const int row = blockIdx.x;
    const int tid = threadIdx.x;
    const float* xr = x + (size_t)row * DM_;
    float xl[4];
    float s = 0.f, q = 0.f;
#pragma unroll
    for (int i = 0; i < 4; ++i) {
        float v = xr[tid + i * 256];
        xl[i] = v; s += v; q += v * v;
    }
    ssum[tid] = s; ssq[tid] = q;
    __syncthreads();
    for (int st = 128; st > 0; st >>= 1) {
        if (tid < st) { ssum[tid] += ssum[tid + st]; ssq[tid] += ssq[tid + st]; }
        __syncthreads();
    }
    const float mu  = ssum[0] * (1.0f / DM_);
    const float var = ssq[0] * (1.0f / DM_) - mu * mu;
    const float rs  = rsqrtf(var + 1e-5f);
    __bf16* orow = out + (size_t)row * DM_;
#pragma unroll
    for (int i = 0; i < 4; ++i) {
        int k = tid + i * 256;
        orow[k] = f2bf((xl[i] - mu) * rs * g[k] + b[k]);
    }
}

// ---------------------------------------------------------------------------
// Sigmoid gates: sd = sigmoid(n@Wsd+bsd), pd = sigmoid(n@Wpd+bpd); [M,8] each
// ---------------------------------------------------------------------------
__global__ __launch_bounds__(64)
void gates_kernel(const __bf16* __restrict__ nbf,
                  const float* __restrict__ Wsd, const float* __restrict__ bsd,
                  const float* __restrict__ Wpd, const float* __restrict__ bpd,
                  float* __restrict__ sdo, float* __restrict__ pdo) {
    __shared__ float red[16][64];
    const int row = blockIdx.x;
    const int tid = threadIdx.x;
    float accs[8] = {0, 0, 0, 0, 0, 0, 0, 0};
    float accp[8] = {0, 0, 0, 0, 0, 0, 0, 0};
    const __bf16* nr = nbf + (size_t)row * DM_;
    for (int k = tid; k < DM_; k += 64) {
        float xv = bf2f(nr[k]);
#pragma unroll
        for (int h = 0; h < 8; ++h) {
            accs[h] += xv * Wsd[k * 8 + h];
            accp[h] += xv * Wpd[k * 8 + h];
        }
    }
#pragma unroll
    for (int h = 0; h < 8; ++h) { red[h][tid] = accs[h]; red[h + 8][tid] = accp[h]; }
    __syncthreads();
    for (int st = 32; st > 0; st >>= 1) {
        if (tid < st) {
#pragma unroll
            for (int j = 0; j < 16; ++j) red[j][tid] += red[j][tid + st];
        }
        __syncthreads();
    }
    if (tid < 8) {
        sdo[(size_t)row * 8 + tid] = 1.0f / (1.0f + expf(-(red[tid][0] + bsd[tid])));
    } else if (tid < 16) {
        int h = tid - 8;
        pdo[(size_t)row * 8 + h] = 1.0f / (1.0f + expf(-(red[tid][0] + bpd[h])));
    }
}

// ---------------------------------------------------------------------------
// Sequential pairwise-state scan. One block per (b,h); pair[64][64] in regs.
// ---------------------------------------------------------------------------
__global__ __launch_bounds__(256)
void scan_kernel(const __bf16* __restrict__ a, const __bf16* __restrict__ bb,
                 const __bf16* __restrict__ ql, const __bf16* __restrict__ qr,
                 const float* __restrict__ sd, const float* __restrict__ pd,
                 __bf16* __restrict__ ys) {
    const int b = blockIdx.x / H_;
    const int h = blockIdx.x % H_;
    const int tid = threadIdx.x;
    const int e  = tid & 63;
    const int dg = tid >> 6;
    const int d0 = dg * 16;

    __shared__ float s_a[64], s_b[64], s_ql[64], s_qr[64];
    __shared__ float s_state[2][64];
    __shared__ float s_part[4][64];

    float pairloc[16];
#pragma unroll
    for (int i = 0; i < 16; ++i) pairloc[i] = 0.f;
    if (tid < 64) s_state[0][tid] = 0.f;
    __syncthreads();

    for (int t = 0; t < T_; ++t) {
        const size_t rowbase = ((size_t)b * T_ + t) * INNER_ + (size_t)h * SD_;
        const size_t gidx    = ((size_t)b * T_ + t) * H_ + h;
        const float sdv = sd[gidx];
        const float pdv = pd[gidx];
        const int p = t & 1;

        if (tid < 64) {
            float av = bf2f(a[rowbase + tid]);
            s_a[tid]  = av;
            s_b[tid]  = bf2f(bb[rowbase + tid]);
            s_ql[tid] = bf2f(ql[rowbase + tid]);
            s_qr[tid] = bf2f(qr[rowbase + tid]);
            s_state[p ^ 1][tid] = sdv * s_state[p][tid] + (1.f - sdv) * av;
        }
        __syncthreads();

        const float be = s_b[e];
        float part = 0.f;
#pragma unroll
        for (int i = 0; i < 16; ++i) {
            float ps = s_state[p][d0 + i];   // prev_state
            pairloc[i] = pdv * pairloc[i] + (1.f - pdv) * ps * be;
            part += pairloc[i] * s_ql[d0 + i];
        }
        s_part[dg][e] = part;
        __syncthreads();

        if (tid < 64) {
            float left = s_part[0][tid] + s_part[1][tid] + s_part[2][tid] + s_part[3][tid];
            ys[rowbase + tid] = f2bf(left * s_qr[tid]);
        }
        __syncthreads();
    }
}

// ---------------------------------------------------------------------------
// WMMA bf16 GEMM:  C[M,N] = A[M,K] @ W[K,N], W supplied TRANSPOSED as
// WT[N][K]. Block tile 64x128, 8 wave32s (2x4), 32x32 per wave = 2x2 frags.
// Double-buffered LDS; async global->LDS staging on gfx1250.
// MODE 0: out_bf16 = tanh(acc)
// MODE 1: out_f32  = acc + resid
// MODE 2: out_bf16 = gelu(acc + bias)
// MODE 3: out_f32  = acc + bias + resid
// M%64==0, N%128==0, K%32==0.
// ---------------------------------------------------------------------------
template <int MODE>
__global__ __launch_bounds__(256)
void gemm_wmma_kernel(const __bf16* __restrict__ Ap, const __bf16* __restrict__ WT,
                      float* __restrict__ outF, __bf16* __restrict__ outB,
                      const float* __restrict__ bias, const float* __restrict__ resid,
                      int N, int K) {
    __shared__ __align__(16) __bf16 As[2][64][32];
    __shared__ __align__(16) __bf16 Bt[2][128][32];   // [n][k]

    const int bm   = blockIdx.x;
    const int bn   = blockIdx.y;
    const int tid  = threadIdx.x;
    const int wid  = tid >> 5;
    const int lane = tid & 31;
    const int lr   = lane & 15;
    const int hi   = lane >> 4;
    const int mBase = (wid >> 2) * 32;
    const int nBase = (wid & 3) * 32;

    // staging coordinates (16B per thread for A; 2x16B for B)
    const int rA = tid >> 2;
    const int cA = (tid & 3) * 8;
    const int rB0 = tid >> 2;            // rows 0..63
    const int rB1 = (tid >> 2) + 64;     // rows 64..127
    const int cB  = (tid & 3) * 8;

    const __bf16* gA  = Ap + (size_t)(bm * 64 + rA) * K + cA;
    const __bf16* gB0 = WT + (size_t)(bn * 128 + rB0) * K + cB;
    const __bf16* gB1 = WT + (size_t)(bn * 128 + rB1) * K + cB;

    v8f acc[2][2];
#pragma unroll
    for (int i = 0; i < 2; ++i)
#pragma unroll
        for (int j = 0; j < 2; ++j) {
            v8f z = {};
            acc[i][j] = z;
        }

    const int NK = K >> 5;

    // ---- stage tile 0 into buffer 0 ----
#if USE_ASYNC_LDS
    ASYNC_COPY16(gA,  &As[0][rA][cA]);
    ASYNC_COPY16(gB0, &Bt[0][rB0][cB]);
    ASYNC_COPY16(gB1, &Bt[0][rB1][cB]);
#else
    *(v8bf*)&As[0][rA][cA]  = *(const v8bf*)gA;
    *(v8bf*)&Bt[0][rB0][cB] = *(const v8bf*)gB0;
    *(v8bf*)&Bt[0][rB1][cB] = *(const v8bf*)gB1;
#endif

    int buf = 0;
    for (int kt = 0; kt < NK; ++kt) {
#if USE_ASYNC_LDS
        wait_async_lds();
#endif
        __syncthreads();   // tile `buf` ready for everyone; buf^1 free

        if (kt + 1 < NK) {
            const size_t ko = (size_t)(kt + 1) * 32;
#if USE_ASYNC_LDS
            ASYNC_COPY16(gA + ko,  &As[buf ^ 1][rA][cA]);
            ASYNC_COPY16(gB0 + ko, &Bt[buf ^ 1][rB0][cB]);
            ASYNC_COPY16(gB1 + ko, &Bt[buf ^ 1][rB1][cB]);
#else
            *(v8bf*)&As[buf ^ 1][rA][cA]  = *(const v8bf*)(gA + ko);
            *(v8bf*)&Bt[buf ^ 1][rB0][cB] = *(const v8bf*)(gB0 + ko);
            *(v8bf*)&Bt[buf ^ 1][rB1][cB] = *(const v8bf*)(gB1 + ko);
#endif
        }

        // ---- fragments (ISA 16-bit A 16x32 / B 32x16 lane layouts) ----
        v16bf afr[2], bfr[2];
#pragma unroll
        for (int f = 0; f < 2; ++f) {
            int row = mBase + f * 16 + lr;
            v8bf lo = *(const v8bf*)&As[buf][row][hi * 8];
            v8bf hh = *(const v8bf*)&As[buf][row][hi * 8 + 16];
#pragma unroll
            for (int i = 0; i < 8; ++i) { afr[f][i] = lo[i]; afr[f][i + 8] = hh[i]; }
            int col = nBase + f * 16 + lr;
            v8bf c0 = *(const v8bf*)&Bt[buf][col][hi * 8];
            v8bf c1 = *(const v8bf*)&Bt[buf][col][hi * 8 + 16];
#pragma unroll
            for (int i = 0; i < 8; ++i) { bfr[f][i] = c0[i]; bfr[f][i + 8] = c1[i]; }
        }
#pragma unroll
        for (int i = 0; i < 2; ++i)
#pragma unroll
            for (int j = 0; j < 2; ++j)
                acc[i][j] = __builtin_amdgcn_wmma_f32_16x16x32_bf16(
                    false, afr[i], false, bfr[j], (short)0, acc[i][j], false, false);

        buf ^= 1;
    }

    // ---- epilogue (C/D layout: row = hi*8+v, col = lr within 16x16 frag) ----
#pragma unroll
    for (int i = 0; i < 2; ++i)
#pragma unroll
        for (int j = 0; j < 2; ++j)
#pragma unroll
            for (int v = 0; v < 8; ++v) {
                int row = bm * 64 + mBase + i * 16 + hi * 8 + v;
                int col = bn * 128 + nBase + j * 16 + lr;
                size_t idx = (size_t)row * N + col;
                float a = acc[i][j][v];
                if (MODE == 0) {
                    outB[idx] = f2bf(tanhf(a));
                } else if (MODE == 1) {
                    outF[idx] = a + resid[idx];
                } else if (MODE == 2) {
                    float vv = a + bias[col];
                    outB[idx] = f2bf(0.5f * vv * (1.0f + erff(vv * 0.70710678118654752f)));
                } else {
                    outF[idx] = a + bias[col] + resid[idx];
                }
            }
}

// ---------------------------------------------------------------------------
// Host launcher
// ---------------------------------------------------------------------------
extern "C" void kernel_launch(void* const* d_in, const int* in_sizes, int n_in,
                              void* d_out, int out_size, void* d_ws, size_t ws_size,
                              hipStream_t stream) {
    (void)in_sizes; (void)n_in; (void)out_size; (void)ws_size;

    const float* x     = (const float*)d_in[0];
    const float* ln_g  = (const float*)d_in[1];
    const float* ln_b  = (const float*)d_in[2];
    const float* ffn_g = (const float*)d_in[3];
    const float* ffn_b = (const float*)d_in[4];
    const float* Wa    = (const float*)d_in[5];
    const float* Wb    = (const float*)d_in[6];
    const float* Wql   = (const float*)d_in[7];
    const float* Wqr   = (const float*)d_in[8];
    const float* Wsd   = (const float*)d_in[9];
    const float* bsd   = (const float*)d_in[10];
    const float* Wpd   = (const float*)d_in[11];
    const float* bpd   = (const float*)d_in[12];
    const float* Wout  = (const float*)d_in[13];
    const float* W1    = (const float*)d_in[14];
    const float* b1    = (const float*)d_in[15];
    const float* W2    = (const float*)d_in[16];
    const float* b2    = (const float*)d_in[17];
    float* out = (float*)d_out;

    char* ws = (char*)d_ws;
    size_t off = 0;
    auto wsalloc = [&](size_t bytes) -> void* {
        void* p = ws + off;
        off = (off + bytes + 255) & ~(size_t)255;
        return p;
    };

    // Transposed bf16 weights: WT[N][K]
    __bf16* wWaT   = (__bf16*)wsalloc((size_t)DM_ * INNER_ * 2);
    __bf16* wWbT   = (__bf16*)wsalloc((size_t)DM_ * INNER_ * 2);
    __bf16* wWqlT  = (__bf16*)wsalloc((size_t)DM_ * INNER_ * 2);
    __bf16* wWqrT  = (__bf16*)wsalloc((size_t)DM_ * INNER_ * 2);
    __bf16* wWoutT = (__bf16*)wsalloc((size_t)INNER_ * DM_ * 2);
    __bf16* wW1T   = (__bf16*)wsalloc((size_t)DM_ * HID_ * 2);
    __bf16* wW2T   = (__bf16*)wsalloc((size_t)HID_ * DM_ * 2);
    __bf16* nbf    = (__bf16*)wsalloc((size_t)M_ * DM_ * 2);     // LN(x); reused as ffn_in
    __bf16* projA  = (__bf16*)wsalloc((size_t)M_ * INNER_ * 2);
    __bf16* projB  = (__bf16*)wsalloc((size_t)M_ * INNER_ * 2);
    __bf16* projQL = (__bf16*)wsalloc((size_t)M_ * INNER_ * 2);
    __bf16* projQR = (__bf16*)wsalloc((size_t)M_ * INNER_ * 2);
    float*  sdbuf  = (float*)wsalloc((size_t)M_ * H_ * 4);
    float*  pdbuf  = (float*)wsalloc((size_t)M_ * H_ * 4);
    __bf16* ysbf   = (__bf16*)wsalloc((size_t)M_ * INNER_ * 2);
    __bf16* actbf  = (__bf16*)wsalloc((size_t)2048 * HID_ * 2);  // FFN chunk buffer

    // 1) weight conversion + transpose: W[K][N] f32 -> WT[N][K] bf16
    {
        dim3 tp(DM_ / 32, INNER_ / 32);
        transpose_to_bf16_kernel<<<tp, 256, 0, stream>>>(Wa,  wWaT,  DM_, INNER_);
        transpose_to_bf16_kernel<<<tp, 256, 0, stream>>>(Wb,  wWbT,  DM_, INNER_);
        transpose_to_bf16_kernel<<<tp, 256, 0, stream>>>(Wql, wWqlT, DM_, INNER_);
        transpose_to_bf16_kernel<<<tp, 256, 0, stream>>>(Wqr, wWqrT, DM_, INNER_);
        dim3 to(INNER_ / 32, DM_ / 32);
        transpose_to_bf16_kernel<<<to, 256, 0, stream>>>(Wout, wWoutT, INNER_, DM_);
        dim3 t1(DM_ / 32, HID_ / 32);
        transpose_to_bf16_kernel<<<t1, 256, 0, stream>>>(W1, wW1T, DM_, HID_);
        dim3 t2(HID_ / 32, DM_ / 32);
        transpose_to_bf16_kernel<<<t2, 256, 0, stream>>>(W2, wW2T, HID_, DM_);
    }

    // 2) layernorm(x) -> nbf
    layernorm_bf16_kernel<<<M_, 256, 0, stream>>>(x, ln_g, ln_b, nbf);

    // 3) sigmoid gates
    gates_kernel<<<M_, 64, 0, stream>>>(nbf, Wsd, bsd, Wpd, bpd, sdbuf, pdbuf);

    // 4) tanh projections via WMMA
    dim3 g1(M_ / 64, INNER_ / 128);
    gemm_wmma_kernel<0><<<g1, 256, 0, stream>>>(nbf, wWaT,  nullptr, projA,  nullptr, nullptr, INNER_, DM_);
    gemm_wmma_kernel<0><<<g1, 256, 0, stream>>>(nbf, wWbT,  nullptr, projB,  nullptr, nullptr, INNER_, DM_);
    gemm_wmma_kernel<0><<<g1, 256, 0, stream>>>(nbf, wWqlT, nullptr, projQL, nullptr, nullptr, INNER_, DM_);
    gemm_wmma_kernel<0><<<g1, 256, 0, stream>>>(nbf, wWqrT, nullptr, projQR, nullptr, nullptr, INNER_, DM_);

    // 5) sequential scan over T (64 independent (b,h) recurrences)
    scan_kernel<<<B_ * H_, 256, 0, stream>>>(projA, projB, projQL, projQR, sdbuf, pdbuf, ysbf);

    // 6) h = x + ys @ Wout  (written to d_out)
    dim3 g2(M_ / 64, DM_ / 128);
    gemm_wmma_kernel<1><<<g2, 256, 0, stream>>>(ysbf, wWoutT, out, nullptr, nullptr, x, DM_, INNER_);

    // 7) ffn_in = layernorm(h) -> nbf (reuse)
    layernorm_bf16_kernel<<<M_, 256, 0, stream>>>(out, ffn_g, ffn_b, nbf);

    // 8) FFN in 2048-row chunks: act = gelu(ffn_in@W1+b1); out += act@W2+b2
    const int CH = 2048;
    dim3 g3(CH / 64, HID_ / 128);
    dim3 g4(CH / 64, DM_ / 128);
    for (int c = 0; c < M_ / CH; ++c) {
        const __bf16* ain = nbf + (size_t)c * CH * DM_;
        float* hout = out + (size_t)c * CH * DM_;
        gemm_wmma_kernel<2><<<g3, 256, 0, stream>>>(ain, wW1T, nullptr, actbf, b1, nullptr, HID_, DM_);
        gemm_wmma_kernel<3><<<g4, 256, 0, stream>>>(actbf, wW2T, hout, nullptr, b2, hout, DM_, HID_);
    }
}
